// BlockLocalAttentionProduct_72121090834679
// MI455X (gfx1250) — matrix operand order repaired and checked by reading, not compiled, not run on hardware
//
#include <hip/hip_runtime.h>
#include <hip/hip_bf16.h>

typedef __bf16 v16bf __attribute__((ext_vector_type(16)));
typedef __bf16 v4bf  __attribute__((ext_vector_type(4)));
typedef float  v8f   __attribute__((ext_vector_type(8)));

namespace {
constexpr int kN = 2, kH = 12, kT = 8192, kD = 64, kG = 64;
constexpr int kHalf = 64, kWin = 192, kKtot = 256;
constexpr int kNB = 2 * kT / 128;                 // 128 blocks per (n,h)

// LDS strides (bf16 elements). 16B-aligned rows, low bank conflicts.
constexpr int QSTR = 72;
constexpr int KSTR = 72;
constexpr int PSTR = 264;
constexpr int VSTR = 264;

constexpr int OFF_Q = 0;
constexpr int OFF_K = OFF_Q + 64 * QSTR * 2;        // 9216
constexpr int OFF_V = OFF_K + kKtot * KSTR * 2;     // 46080
constexpr int SMEM_BYTES = OFF_V + kD * VSTR * 2;   // 79872 (~78KB -> 4 WGs/WGP)
} // namespace

// ---- WMMA helper (CDNA5 16x16x32 bf16, f32 accumulate) ----
__device__ __forceinline__ v8f wmma_bf16(v16bf a, v16bf b, v8f c) {
  return __builtin_amdgcn_wmma_f32_16x16x32_bf16(false, a, false, b,
                                                 (short)0, c, false, false);
}

// A-matrix 16x32 bf16 fragment from row-major LDS.
__device__ __forceinline__ v16bf load_a(const __bf16* base, int stride,
                                        int mbase, int kbase, int lane) {
  int m    = mbase + (lane & 15);
  int ksel = (lane >> 4) << 3;                 // 0 or 8
  const __bf16* p = base + m * stride + kbase + ksel;
  v16bf a;
  reinterpret_cast<uint4*>(&a)[0] = *reinterpret_cast<const uint4*>(p);
  reinterpret_cast<uint4*>(&a)[1] = *reinterpret_cast<const uint4*>(p + 16);
  return a;
}

// B-matrix 32x16 bf16 fragment: lane = column N, K in table order (8h..8h+7, 16+8h..16+8h+7).
__device__ __forceinline__ v16bf load_b(const __bf16* base, int stride,
                                        int nbase, int kbase, int lane) {
  int n  = nbase + (lane & 15);
  int kk = kbase + ((lane >> 4) << 3);         // 0 or 8
  const __bf16* p = base + n * stride + kk;
  v16bf b;
  reinterpret_cast<uint4*>(&b)[0] = *reinterpret_cast<const uint4*>(p);
  reinterpret_cast<uint4*>(&b)[1] = *reinterpret_cast<const uint4*>(p + 16);
  return b;
}

__device__ __forceinline__ v4bf pack_bf4(float4 f) {
  v4bf o;
  o[0] = (__bf16)f.x; o[1] = (__bf16)f.y;
  o[2] = (__bf16)f.z; o[3] = (__bf16)f.w;
  return o;
}

__global__ void __launch_bounds__(128)
block_local_attn_wmma(const float* __restrict__ q,
                      const float* __restrict__ k,
                      const float* __restrict__ v,
                      const float* __restrict__ am,
                      const float* __restrict__ gk,
                      const float* __restrict__ gv,
                      const float* __restrict__ gm,
                      float* __restrict__ out) {
  extern __shared__ unsigned char smem[];
  __bf16* Qs = reinterpret_cast<__bf16*>(smem + OFF_Q);
  __bf16* Ks = reinterpret_cast<__bf16*>(smem + OFF_K);
  __bf16* Ps = Ks;   // aliased: K dead after phase 1, barrier-separated
  __bf16* Vt = reinterpret_cast<__bf16*>(smem + OFF_V);

  const int tid  = threadIdx.x;
  const int lane = tid & 31;
  const int wave = tid >> 5;

  const int blk   = blockIdx.x;            // 0 .. N*H*NB-1
  const int nh    = blk / kNB;
  const int b     = blk - nh * kNB;
  const int nidx  = nh / kH;
  const int kwin0 = b * kHalf - kHalf;     // first key index of window

  const size_t qkv_base = (size_t)nh * kT * kD;
  const size_t g_base   = (size_t)nh * kG * kD;

  const int mbase = wave * 16;             // 16 query rows per wave
  const int mrow0 = mbase + ((lane >> 4) << 3);
  const int ncl   = lane & 15;

  // ---- Mask column values (column j = nt*16 + ncl; row-independent). ----
  float mcol[16];
#pragma unroll
  for (int nt = 0; nt < 16; ++nt) {
    int j = nt * 16 + ncl;
    if (j < kWin) {                        // compile-time branch after unroll
      int ki = kwin0 + j;
      mcol[nt] = (ki >= 0 && ki < kT) ? am[(size_t)nidx * kT + ki] : -10000.0f;
    } else {
      mcol[nt] = gm[(size_t)nidx * kG + (j - kWin)];
    }
  }

  // ---- Stage Q tile: float4 loads -> packed bf16x4 LDS stores ----
  for (int i = tid; i < 64 * (kD / 4); i += 128) {
    int r = i >> 4, c4 = (i & 15) << 2;
    float4 qv = *reinterpret_cast<const float4*>(
        &q[qkv_base + (size_t)(b * kHalf + r) * kD + c4]);
    *reinterpret_cast<v4bf*>(&Qs[r * QSTR + c4]) = pack_bf4(qv);
  }
  // ---- Stage K window+globals (row-major) and V (transposed Vt[d][k]) ----
  for (int i = tid; i < kKtot * (kD / 4); i += 128) {
    int j = i >> 4, c4 = (i & 15) << 2;    // j = key slot, c4 = d offset
    float4 kv4 = make_float4(0.f, 0.f, 0.f, 0.f);
    float4 vv4 = make_float4(0.f, 0.f, 0.f, 0.f);
    if (j < kWin) {
      int ki = kwin0 + j;
      if (ki >= 0 && ki < kT) {
        kv4 = *reinterpret_cast<const float4*>(&k[qkv_base + (size_t)ki * kD + c4]);
        vv4 = *reinterpret_cast<const float4*>(&v[qkv_base + (size_t)ki * kD + c4]);
      }
    } else {
      int jg = j - kWin;
      kv4 = *reinterpret_cast<const float4*>(&gk[g_base + (size_t)jg * kD + c4]);
      vv4 = *reinterpret_cast<const float4*>(&gv[g_base + (size_t)jg * kD + c4]);
    }
    *reinterpret_cast<v4bf*>(&Ks[j * KSTR + c4]) = pack_bf4(kv4);
    Vt[(c4 + 0) * VSTR + j] = (__bf16)vv4.x;
    Vt[(c4 + 1) * VSTR + j] = (__bf16)vv4.y;
    Vt[(c4 + 2) * VSTR + j] = (__bf16)vv4.z;
    Vt[(c4 + 3) * VSTR + j] = (__bf16)vv4.w;
  }
  __syncthreads();

  // ---- Phase 1: S = Q K^T, scores stay in accumulators (16 N-tiles) ----
  v8f acc[16];
  {
    v16bf a0 = load_a(Qs, QSTR, mbase, 0,  lane);
    v16bf a1 = load_a(Qs, QSTR, mbase, 32, lane);
    v16bf b0 = load_b(Ks, KSTR, 0, 0,  lane);
    v16bf b1 = load_b(Ks, KSTR, 0, 32, lane);
#pragma unroll
    for (int nt = 0; nt < 16; ++nt) {
      v16bf c0 = b0, c1 = b1;
      if (nt < 15) {                       // prefetch next tile's B frags
        b0 = load_b(Ks, KSTR, (nt + 1) * 16, 0,  lane);
        b1 = load_b(Ks, KSTR, (nt + 1) * 16, 32, lane);
      }
      v8f c = {0.f, 0.f, 0.f, 0.f, 0.f, 0.f, 0.f, 0.f};
      c = wmma_bf16(a0, c0, c);
      c = wmma_bf16(a1, c1, c);
      acc[nt] = c;
    }
  }

  // ---- Phase 2: register softmax (unnormalized; 1/sum folded into epilogue).
  //      Row mrow0+e lives at acc[nt][e] across the 16 lanes of this half. ----
  float rmax[8], rsum[8], rinv[8];
#pragma unroll
  for (int e = 0; e < 8; ++e) rmax[e] = -3.0e38f;
#pragma unroll
  for (int nt = 0; nt < 16; ++nt) {
#pragma unroll
    for (int e = 0; e < 8; ++e) {
      float x = acc[nt][e] * 0.125f + mcol[nt];   // 1/sqrt(64) + mask
      acc[nt][e] = x;
      rmax[e] = fmaxf(rmax[e], x);
    }
  }
#pragma unroll
  for (int off = 1; off <= 8; off <<= 1)
#pragma unroll
    for (int e = 0; e < 8; ++e)
      rmax[e] = fmaxf(rmax[e], __shfl_xor(rmax[e], off, 32));

#pragma unroll
  for (int e = 0; e < 8; ++e) rsum[e] = 0.f;
#pragma unroll
  for (int nt = 0; nt < 16; ++nt) {
#pragma unroll
    for (int e = 0; e < 8; ++e) {
      float p = __expf(acc[nt][e] - rmax[e]);
      acc[nt][e] = p;
      rsum[e] += p;
    }
  }
#pragma unroll
  for (int off = 1; off <= 8; off <<= 1)
#pragma unroll
    for (int e = 0; e < 8; ++e)
      rsum[e] += __shfl_xor(rsum[e], off, 32);
#pragma unroll
  for (int e = 0; e < 8; ++e) rinv[e] = 1.0f / rsum[e];

  __syncthreads();                         // all waves done reading Ks
  // Write unnormalized P (bf16) into Ps (aliased over Ks) for A-relayout.
#pragma unroll
  for (int nt = 0; nt < 16; ++nt) {
#pragma unroll
    for (int e = 0; e < 8; ++e)
      Ps[(mrow0 + e) * PSTR + nt * 16 + ncl] = (__bf16)acc[nt][e];
  }
  __syncthreads();

  // ---- Phase 3: ctx = (P V) * rinv (8 K-steps of 32 per output tile) ----
  {
    v16bf af[8];
#pragma unroll
    for (int kk = 0; kk < 8; ++kk)
      af[kk] = load_a(Ps, PSTR, mbase, kk * 32, lane);

    float* orow = out + qkv_base + (size_t)(b * kHalf) * kD;
#pragma unroll
    for (int dt = 0; dt < 4; ++dt) {
      v8f c = {0.f, 0.f, 0.f, 0.f, 0.f, 0.f, 0.f, 0.f};
#pragma unroll
      for (int kk = 0; kk < 8; ++kk) {
        v16bf bb = load_b(Vt, VSTR, dt * 16, kk * 32, lane);
        c = wmma_bf16(af[kk], bb, c);
      }
      int ncol = dt * 16 + ncl;
#pragma unroll
      for (int e = 0; e < 8; ++e)
        orow[(size_t)(mrow0 + e) * kD + ncol] = c[e] * rinv[e];
    }
  }
}

extern "C" void kernel_launch(void* const* d_in, const int* in_sizes, int n_in,
                              void* d_out, int out_size, void* d_ws, size_t ws_size,
                              hipStream_t stream) {
  const float* q  = (const float*)d_in[0];   // query_layer  (n,h,t,d)
  const float* k  = (const float*)d_in[1];   // key_layer    (n,h,t,d)
  const float* v  = (const float*)d_in[2];   // value_layer  (n,h,t,d)
  const float* am = (const float*)d_in[3];   // attention_mask (n,1,1,t)
  const float* gk = (const float*)d_in[4];   // global_key   (n,h,g,d)
  const float* gv = (const float*)d_in[5];   // global_value (n,h,g,d)
  const float* gm = (const float*)d_in[6];   // global_mask  (n,1,1,g)
  float* out = (float*)d_out;

  dim3 grid(kN * kH * kNB);                  // 3072 tiles, one WG each
  dim3 block(128);                           // 4 wave32
  block_local_attn_wmma<<<grid, block, SMEM_BYTES, stream>>>(
      q, k, v, am, gk, gv, gm, out);
}
